// CrossViewAttention_61899068670540
// MI455X (gfx1250) — compile-verified, matrix operand order checked by reading, not run on hardware
//
#include <hip/hip_runtime.h>
#include <cstdint>
#include <cstddef>

// ---------------- problem constants ----------------
#define CB    2          // batch
#define CV    8          // views
#define CS    256        // sequence per view
#define CD    2048       // model dim
#define CNKV  8          // kv heads
#define GRP   4          // query heads per kv head
#define HDIM  64         // head dim
#define KVW   1024       // 2*NKV*HDIM
#define MROWS 4096       // CB*CV*CS

typedef __attribute__((ext_vector_type(16))) __bf16 v16bf;
typedef __attribute__((ext_vector_type(8)))  __bf16 v8bf;
typedef __attribute__((ext_vector_type(8)))  float  v8f;

static_assert(sizeof(__bf16) == 2, "bf16 size");

// ---------------- scalar helpers ----------------
__device__ __forceinline__ unsigned short bf16_bits(float f) {
  unsigned int u = __builtin_bit_cast(unsigned int, f);
  return (unsigned short)((u + 0x7FFFu + ((u >> 16) & 1u)) >> 16);
}
__device__ __forceinline__ __bf16 f32_to_bf16(float f) {
  unsigned short h = bf16_bits(f);
  return __builtin_bit_cast(__bf16, h);
}
__device__ __forceinline__ void store_out(float* p, float v)  { *p = v; }
__device__ __forceinline__ void store_out(__bf16* p, float v) { *p = f32_to_bf16(v); }

// ---- CDNA5 async global->LDS copy (ASYNCcnt-tracked DMA, no VGPR staging) ----
__device__ __forceinline__ void async_load_b128(const void* g, void* l) {
  asm volatile("global_load_async_to_lds_b128 %0, %1, off"
               :: "v"((unsigned)(uintptr_t)l), "v"(g)
               : "memory");
}
__device__ __forceinline__ void wait_async0() {
  asm volatile("s_wait_asynccnt 0x0" ::: "memory");
}

// ---- VALU cross-lane xor butterfly within 16-lane rows (v_permlane16_b32) ----
// Keeps softmax reductions off the LDS pipe (no ds_bpermute).
template <int MASK, unsigned S0, unsigned S1>
__device__ __forceinline__ float lane_xor16(float x) {
#if __has_builtin(__builtin_amdgcn_permlane16)
  unsigned u = __builtin_bit_cast(unsigned, x);
  unsigned r = __builtin_amdgcn_permlane16(u, u, S0, S1, false, false);
  return __builtin_bit_cast(float, r);
#else
  return __shfl_xor(x, MASK, 32);
#endif
}
__device__ __forceinline__ float red_max16(float m) {
  m = fmaxf(m, lane_xor16<1, 0x67452301u, 0xEFCDAB89u>(m));
  m = fmaxf(m, lane_xor16<2, 0x54761032u, 0xDCFE98BAu>(m));
  m = fmaxf(m, lane_xor16<4, 0x32107654u, 0xBA98FEDCu>(m));
  m = fmaxf(m, lane_xor16<8, 0xFEDCBA98u, 0x76543210u>(m));
  return m;
}
__device__ __forceinline__ float red_sum16(float s) {
  s += lane_xor16<1, 0x67452301u, 0xEFCDAB89u>(s);
  s += lane_xor16<2, 0x54761032u, 0xDCFE98BAu>(s);
  s += lane_xor16<4, 0x32107654u, 0xBA98FEDCu>(s);
  s += lane_xor16<8, 0xFEDCBA98u, 0x76543210u>(s);
  return s;
}

// Fragment-native tile: a 32(K)x16 operand subtile stored as 1KB block:
//   [part(2: elems 0..7 / 8..15)][lane(32)][8 bf16]
// lane = half*16 + cross, where cross = N-col (B) or M-row (A) and half picks
// contraction elems {kh..kh+7, kh+16..kh+23}, kh = half*8 (ISA 16-bit layouts).
__device__ __forceinline__ void frag_pos(int kl, int& half, int& e) {
  half = (kl >> 3) & 1;
  e = (kl & 7) + ((kl & 16) ? 8 : 0);
}
__device__ __forceinline__ size_t frag_off(int half, int cross, int e) {
  return (size_t)((e >= 8) ? 256 : 0) + (size_t)(half * 16 + cross) * 8 + (e & 7);
}
// contiguous 2 x b128 per lane
__device__ __forceinline__ v16bf load_frag(const __bf16* tile) {
  const int lane = threadIdx.x & 31;
  v8bf lo = *reinterpret_cast<const v8bf*>(tile + lane * 8);
  v8bf hi = *reinterpret_cast<const v8bf*>(tile + 256 + lane * 8);
  v16bf r;
#pragma unroll
  for (int e = 0; e < 8; ++e) { r[e] = lo[e]; r[8 + e] = hi[e]; }
  return r;
}

// A fragment (16x32, MxK) from row-major LDS tile: 2 x b128 per lane.
__device__ __forceinline__ v16bf lds_load_A(const __bf16* base, int ld) {
  const int lane = threadIdx.x & 31;
  const int row  = lane & 15;
  const int kh   = (lane >> 4) << 3;
  const __bf16* p = base + row * ld + kh;
  v8bf lo = *reinterpret_cast<const v8bf*>(p);
  v8bf hi = *reinterpret_cast<const v8bf*>(p + 16);
  v16bf a;
#pragma unroll
  for (int e = 0; e < 8; ++e) { a[e] = lo[e]; a[8 + e] = hi[e]; }
  return a;
}

// ---------------- fp32 -> bf16 row-major convert (x activations) ----------------
__global__ __launch_bounds__(256) void cvt_f32_bf16(const float* __restrict__ in,
                                                    __bf16* __restrict__ out, int n) {
  int i = (blockIdx.x * 256 + threadIdx.x) * 4;
  if (i + 3 < n) {
    float4 f = *reinterpret_cast<const float4*>(in + i);
    uint2 u;
    u.x = (unsigned)bf16_bits(f.x) | ((unsigned)bf16_bits(f.y) << 16);
    u.y = (unsigned)bf16_bits(f.z) | ((unsigned)bf16_bits(f.w) << 16);
    *reinterpret_cast<uint2*>(out + i) = u;
  }
}

// ---------------- fp32 weights -> fragment-native swizzled bf16 ----------------
// W [K][N] row-major f32 -> tiles (nt = n/16, kt = k/32), tile index nt*(K/32)+kt.
__global__ __launch_bounds__(256) void swz_weight(const float* __restrict__ W,
                                                  __bf16* __restrict__ Wswz,
                                                  int K, int N) {
  const int t    = blockIdx.x * 256 + threadIdx.x;
  const int lane = t & 31;
  const int tile = t >> 5;
  const int nkt  = K >> 5;
  const int nt   = tile / nkt;
  const int kt   = tile - nt * nkt;
  const int ncol = nt * 16 + (lane & 15);
  const int k0   = kt * 32 + ((lane >> 4) << 3);
  uint4 lo, hi;
  unsigned* lp = reinterpret_cast<unsigned*>(&lo);
  unsigned* hp = reinterpret_cast<unsigned*>(&hi);
#pragma unroll
  for (int p = 0; p < 4; ++p) {
    lp[p] = (unsigned)bf16_bits(W[(size_t)(k0 + 2 * p) * N + ncol]) |
            ((unsigned)bf16_bits(W[(size_t)(k0 + 2 * p + 1) * N + ncol]) << 16);
    hp[p] = (unsigned)bf16_bits(W[(size_t)(k0 + 16 + 2 * p) * N + ncol]) |
            ((unsigned)bf16_bits(W[(size_t)(k0 + 16 + 2 * p + 1) * N + ncol]) << 16);
  }
  *reinterpret_cast<uint4*>(Wswz + (size_t)tile * 512 + lane * 8) = lo;
  *reinterpret_cast<uint4*>(Wswz + (size_t)tile * 512 + 256 + lane * 8) = hi;
}

// ---------------- GEMM epilogue store modes ----------------
// MODE 0: row-major OutT.  MODE 1: KV -> K/V fragment-swizzled global (C=K, C2=V).
// MODE 2: Q -> fragment-swizzled global.
template <typename OutT, int MODE>
__device__ __forceinline__ void epi_store(OutT* C, __bf16* C2, int N, int r, int cn, float val) {
  if constexpr (MODE == 0) {
    store_out(&C[(size_t)r * N + cn], val);
  } else if constexpr (MODE == 1) {
    const int bv = r >> 8, s = r & 255;
    if (cn < 512) {                       // K half: contract dim = h, cross = key(s)
      const int n = cn >> 6, h = cn & 63;
      int half, e; frag_pos(h & 31, half, e);
      size_t base = ((((size_t)bv * CNKV + n) * 16 + (s >> 4)) * 2 + (h >> 5)) * 512;
      reinterpret_cast<__bf16*>(C)[base + frag_off(half, s & 15, e)] = f32_to_bf16(val);
    } else {                              // V half: contract dim = key(s), cross = h
      const int cc = cn - 512;
      const int n = cc >> 6, h = cc & 63;
      int half, e; frag_pos(s & 31, half, e);
      size_t base = ((((size_t)bv * CNKV + n) * 8 + (s >> 5)) * 4 + (h >> 4)) * 512;
      C2[base + frag_off(half, h & 15, e)] = f32_to_bf16(val);
    }
  } else {                                // Q: A-layout, contract = h, cross = row(s)
    const int bv = r >> 8, s = r & 255;
    const int nq = cn >> 6, h = cn & 63;
    int half, e; frag_pos(h & 31, half, e);
    size_t base = ((((size_t)bv * 32 + nq) * 16 + (s >> 4)) * 2 + (h >> 5)) * 512;
    reinterpret_cast<__bf16*>(C)[base + frag_off(half, s & 15, e)] = f32_to_bf16(val);
  }
}

// ---------------- tiled WMMA GEMM: C = A[M,K] (row-major) * Bswz[K,N] ----------------
// Double-buffered LDS fed by async global->LDS DMA, overlapped with WMMA compute.
#define BM 128
#define BN 128
#define BK 32
#define ALD (BK + 8)   // 40 bf16 rows (80B, 16B-aligned, conflict-minimal)

template <typename OutT, int MODE>
__global__ __launch_bounds__(256) void gemm_ws(const __bf16* __restrict__ A,
                                               const __bf16* __restrict__ Bswz,
                                               OutT* __restrict__ C,
                                               __bf16* __restrict__ C2,
                                               int M, int N, int K) {
  __shared__ __bf16 As[2][BM][ALD];    // 2 x 10KB
  __shared__ __bf16 Bs[2][8 * 512];    // 2 x 8KB fragment tiles

  const int nkt = K >> 5;
  const int nbx = N / BN;
  const int n0  = (blockIdx.x % nbx) * BN;
  const int m0  = (blockIdx.x / nbx) * BM;
  const int nt0 = n0 >> 4;

  const int tid  = threadIdx.x;
  const int lane = tid & 31;
  const int wave = tid >> 5;
  const int wm   = wave & 1;   // 2 waves in M (64 rows each)
  const int wn   = wave >> 1;  // 4 waves in N (32 cols each)

  auto issue_tiles = [&](int kt, int buf) {
#pragma unroll
    for (int it = 0; it < 2; ++it) {   // A tile: 128x32 bf16 = 512 xb128
      int u = tid + it * 256;
      int row = u >> 2, quad = u & 3;
      async_load_b128(A + (size_t)(m0 + row) * K + (size_t)kt * BK + quad * 8,
                      &As[buf][row][quad * 8]);
    }
#pragma unroll
    for (int it = 0; it < 2; ++it) {   // B: 8 fragment tiles (1KB each), dense
      int u = tid + it * 256;
      int j = u >> 6, q = u & 63;
      async_load_b128(Bswz + ((size_t)(nt0 + j) * nkt + kt) * 512 + q * 8,
                      Bs[buf] + u * 8);
    }
  };

  v8f acc[4][2];
#pragma unroll
  for (int ti = 0; ti < 4; ++ti)
#pragma unroll
    for (int tj = 0; tj < 2; ++tj)
#pragma unroll
      for (int e = 0; e < 8; ++e) acc[ti][tj][e] = 0.0f;

  issue_tiles(0, 0);
  wait_async0();
  __syncthreads();

  for (int kt = 0; kt < nkt; ++kt) {
    const int cur = kt & 1;
    if (kt + 1 < nkt) issue_tiles(kt + 1, cur ^ 1);   // overlap DMA with compute

    v16bf bfrag[2];
#pragma unroll
    for (int tj = 0; tj < 2; ++tj)
      bfrag[tj] = load_frag(Bs[cur] + (wn * 2 + tj) * 512);
#pragma unroll
    for (int ti = 0; ti < 4; ++ti) {
      v16bf afrag = lds_load_A(&As[cur][wm * 64 + ti * 16][0], ALD);
#pragma unroll
      for (int tj = 0; tj < 2; ++tj)
        acc[ti][tj] = __builtin_amdgcn_wmma_f32_16x16x32_bf16(
            false, afrag, false, bfrag[tj], (short)0, acc[ti][tj], false, false);
    }

    wait_async0();      // next buffer resident in LDS
    __syncthreads();    // all waves done reading cur, next buffer published
  }

  const int ncol = lane & 15;
  const int mb   = (lane >> 4) << 3;
#pragma unroll
  for (int ti = 0; ti < 4; ++ti)
#pragma unroll
    for (int tj = 0; tj < 2; ++tj)
#pragma unroll
      for (int e = 0; e < 8; ++e) {
        int r  = m0 + wm * 64 + ti * 16 + mb + e;
        int cn = n0 + wn * 32 + tj * 16 + ncol;
        epi_store<OutT, MODE>(C, C2, N, r, cn, acc[ti][tj][e]);
      }
}

// ---------------- fused GQA flash attention ----------------
// Block = (b, v, kv-head n, 128-query tile); 8 waves x 16 query rows.
// Q/K/V pre-swizzled into fragment-native tiles -> all WMMA operands are 2xb128.
// K/V chunks double-buffered via async global->LDS DMA.
__global__ __launch_bounds__(256) void attn_kernel(const __bf16* __restrict__ Qswz,
                                                   const __bf16* __restrict__ Kswz,
                                                   const __bf16* __restrict__ Vswz,
                                                   __bf16* __restrict__ Oout) {
  __shared__ __bf16 Klds[2][8 * 512];  // chunk: 4 key-subtiles x 2 dim-subtiles
  __shared__ __bf16 Vlds[2][8 * 512];  // chunk: 2 key-subtiles x 4 dim-subtiles
  __shared__ __bf16 Plds[8 * 1024];    // per-wave P in A-fragment layout

  const int tid  = threadIdx.x;
  const int lane = tid & 31;
  const int w    = tid >> 5;

  const int bid = blockIdx.x;        // [b][v(3)][n(3)][qt(3)]
  const int qt  = bid & 7;
  const int n   = (bid >> 3) & (CNKV - 1);
  const int v   = (bid >> 6) & (CV - 1);
  const int b   = bid >> 9;
  const int vprev = (v + CV - 1) & (CV - 1);
  const int vnext = (v + 1) & (CV - 1);

  const int rbase = qt * 128 + w * 16;   // query rows r = g*256 + s
  const int ncol  = lane & 15;
  const int mb    = (lane >> 4) << 3;

  auto issue_chunk = [&](int c, int buf) {
    const int cl = (c < 4) ? c : c - 4;
    const int vs = (c < 4) ? vprev : vnext;
    const __bf16* Ksrc =
        Kswz + ((((size_t)(b * CV + vs) * CNKV + n) * 16 + cl * 4) * 2) * 512;
    const __bf16* Vsrc =
        Vswz + ((((size_t)(b * CV + vs) * CNKV + n) * 8 + cl * 2) * 4) * 512;
#pragma unroll
    for (int it = 0; it < 2; ++it) {   // 8KB + 8KB dense async copies
      int u = tid + it * 256;
      async_load_b128(Ksrc + u * 8, Klds[buf] + u * 8);
      async_load_b128(Vsrc + u * 8, Vlds[buf] + u * 8);
    }
  };

  // Q fragments: contiguous swizzled global loads, kept in registers
  v16bf qfrag[2];
  {
    const int g  = rbase >> 8;
    const int s0 = rbase & 255;
    const __bf16* qb =
        Qswz + ((((size_t)(b * CV + v) * 32) + (n * GRP + g)) * 16 + (s0 >> 4)) * 2 * 512;
    qfrag[0] = load_frag(qb);
    qfrag[1] = load_frag(qb + 512);
  }

  float m_run[8], l_run[8];
  v8f o_acc[4];
#pragma unroll
  for (int i = 0; i < 8; ++i) { m_run[i] = -3.0e38f; l_run[i] = 0.0f; }
#pragma unroll
  for (int t = 0; t < 4; ++t)
#pragma unroll
    for (int e = 0; e < 8; ++e) o_acc[t][e] = 0.0f;

  const float scale = 0.125f;  // 1/sqrt(64)

  issue_chunk(0, 0);
  wait_async0();
  __syncthreads();

  for (int c = 0; c < 8; ++c) {     // 8 chunks x 64 keys
    const int cur = c & 1;
    if (c + 1 < 8) issue_chunk(c + 1, cur ^ 1);   // overlap DMA with softmax/WMMA

    // ---- scores: 4 key-subtiles x (2 dim-subtiles of 32) ----
    v8f sacc[4];
#pragma unroll
    for (int t = 0; t < 4; ++t)
#pragma unroll
      for (int e = 0; e < 8; ++e) sacc[t][e] = 0.0f;
#pragma unroll
    for (int tj = 0; tj < 4; ++tj)
#pragma unroll
      for (int kk = 0; kk < 2; ++kk) {
        v16bf kf = load_frag(Klds[cur] + (tj * 2 + kk) * 512);
        sacc[tj] = __builtin_amdgcn_wmma_f32_16x16x32_bf16(
            false, qfrag[kk], false, kf, (short)0, sacc[tj], false, false);
      }

    // ---- online softmax (slot i = row mb+i): VALU permlane16 butterflies ----
    float mnew[8], corr[8], lloc[8];
#pragma unroll
    for (int i = 0; i < 8; ++i) {
      float m = fmaxf(fmaxf(sacc[0][i], sacc[1][i]), fmaxf(sacc[2][i], sacc[3][i]));
      m = red_max16(m);
      mnew[i] = fmaxf(m_run[i], m * scale);
      corr[i] = __expf(m_run[i] - mnew[i]);
      lloc[i] = 0.0f;
    }
#pragma unroll
    for (int tj = 0; tj < 4; ++tj)
#pragma unroll
      for (int i = 0; i < 8; ++i) {
        float p = __expf(sacc[tj][i] * scale - mnew[i]);
        lloc[i] += p;
        int t = tj * 16 + ncol;            // chunk-local key of this value
        int half, e; frag_pos(t & 31, half, e);
        Plds[w * 1024 + (t >> 5) * 512 + frag_off(half, mb + i, e)] = f32_to_bf16(p);
      }
#pragma unroll
    for (int i = 0; i < 8; ++i) {
      lloc[i] = red_sum16(lloc[i]);
      l_run[i] = l_run[i] * corr[i] + lloc[i];
      m_run[i] = mnew[i];
    }
#pragma unroll
    for (int t = 0; t < 4; ++t)
#pragma unroll
      for (int i = 0; i < 8; ++i) o_acc[t][i] *= corr[i];

    // ---- O += P(16x64) @ V(64x64): same-wave LDS RAW handled by DScnt ordering ----
#pragma unroll
    for (int kk = 0; kk < 2; ++kk) {
      v16bf pf = load_frag(Plds + w * 1024 + kk * 512);
#pragma unroll
      for (int tn = 0; tn < 4; ++tn) {
        v16bf vf = load_frag(Vlds[cur] + (kk * 4 + tn) * 512);
        o_acc[tn] = __builtin_amdgcn_wmma_f32_16x16x32_bf16(
            false, pf, false, vf, (short)0, o_acc[tn], false, false);
      }
    }

    wait_async0();      // next chunk resident
    __syncthreads();    // all waves done with cur
  }

  // ---- normalize + write attn output (row-major bf16: feeds final GEMM A-side) ----
#pragma unroll
  for (int tn = 0; tn < 4; ++tn)
#pragma unroll
    for (int i = 0; i < 8; ++i) {
      int r  = rbase + mb + i;
      int g  = r >> 8;
      int sq = r & (CS - 1);
      size_t col = (size_t)(n * GRP + g) * HDIM + tn * 16 + ncol;
      Oout[((size_t)((b * CV + v) * CS + sq)) * CD + col] =
          f32_to_bf16(o_acc[tn][i] / l_run[i]);
    }
}

// ---------------- host orchestration ----------------
extern "C" void kernel_launch(void* const* d_in, const int* in_sizes, int n_in,
                              void* d_out, int out_size, void* d_ws, size_t ws_size,
                              hipStream_t stream) {
  (void)in_sizes; (void)n_in; (void)out_size; (void)ws_size;
  const float* x   = (const float*)d_in[0];
  const float* wq  = (const float*)d_in[1];
  const float* wkv = (const float*)d_in[2];
  const float* wo  = (const float*)d_in[3];
  float* out = (float*)d_out;

  char* ws = (char*)d_ws;
  size_t off = 0;
  auto carve = [&](size_t bytes) {
    char* p = ws + off;
    off += (bytes + 255) & ~(size_t)255;
    return p;
  };
  __bf16* xb    = (__bf16*)carve((size_t)MROWS * CD * 2);       // 16MB row-major x
  __bf16* wqs   = (__bf16*)carve((size_t)CD * CD * 2);          // 8MB swizzled
  __bf16* wkvs  = (__bf16*)carve((size_t)CD * KVW * 2);         // 4MB swizzled
  __bf16* wos   = (__bf16*)carve((size_t)CD * CD * 2);          // 8MB swizzled
  __bf16* qswz  = (__bf16*)carve((size_t)MROWS * CD * 2);       // 16MB Q fragments
  __bf16* kswz  = (__bf16*)carve((size_t)16 * CNKV * 16 * 2 * 512 * 2); // 4MB K fragments
  __bf16* vswz  = (__bf16*)carve((size_t)16 * CNKV * 8 * 4 * 512 * 2);  // 4MB V fragments
  __bf16* attnb = (__bf16*)carve((size_t)MROWS * CD * 2);       // 16MB attn out

  const int nx = MROWS * CD;  // 8388608
  cvt_f32_bf16<<<nx / 1024, 256, 0, stream>>>(x, xb, nx);
  // weight swizzles: threads = K*N/16
  swz_weight<<<(CD * CD / 16) / 256, 256, 0, stream>>>(wq, wqs, CD, CD);
  swz_weight<<<(CD * KVW / 16) / 256, 256, 0, stream>>>(wkv, wkvs, CD, KVW);
  swz_weight<<<(CD * CD / 16) / 256, 256, 0, stream>>>(wo, wos, CD, CD);

  // Q = X@Wq -> fragment-swizzled
  gemm_ws<__bf16, 2><<<(MROWS / BM) * (CD / BN), 256, 0, stream>>>(
      xb, wqs, qswz, nullptr, MROWS, CD, CD);
  // KV = X@Wkv -> K/V fragment-swizzled
  gemm_ws<__bf16, 1><<<(MROWS / BM) * (KVW / BN), 256, 0, stream>>>(
      xb, wkvs, kswz, vswz, MROWS, KVW, CD);
  // fused GQA attention
  attn_kernel<<<CB * CV * CNKV * 8, 256, 0, stream>>>(qswz, kswz, vswz, attnb);
  // OUT = Attn@Wo (fp32)
  gemm_ws<float, 0><<<(MROWS / BM) * (CD / BN), 256, 0, stream>>>(
      attnb, wos, out, nullptr, MROWS, CD, CD);
}
